// FusionGAT_1906965479907
// MI455X (gfx1250) — matrix operand order repaired and checked by reading, not compiled, not run on hardware
//
#include <hip/hip_runtime.h>
#include <cstdint>
#include <cstddef>

typedef __attribute__((ext_vector_type(16))) _Float16 v16h;
typedef __attribute__((ext_vector_type(8)))  float    v8f;

#define NTOT   3072
#define FIN    512
#define NHID   64
#define NHEADS 4
#define NGRAPH 3
#define NCLASS 16
#define GH     (NGRAPH*NHEADS)   /* 12 */
#define ALPHAC 0.2f

static __device__ __forceinline__ v8f wmma_f16(v16h a, v16h b, v8f c) {
  // D = A(16x32 f16) * B(32x16 f16) + C(16x16 f32)
  return __builtin_amdgcn_wmma_f32_16x16x32_f16(false, a, false, b, (short)0, c,
                                                false, false);
}

// A-operand slot j of lane l maps to K = j + 16*(l/16), row M = l%16
// B-operand slot j of lane l maps to K = j + 16*(l/16), col N = l%16
// C VGPR r of lane l maps to row M = r + 8*(l/16), col N = l%16

// ---------------------------------------------------------------------------
// Kernel 1: Wh = x @ W per (g,h); also f1 = Wh.a1, f2 = Wh.a2.
// Emits WhT[gh][o][n] (f16, B-load-ready for kernel 2) and f1/f2 (f32).
// ---------------------------------------------------------------------------
__global__ void __launch_bounds__(256)
wh_kernel(const float* __restrict__ x, const float* __restrict__ W,
          const float* __restrict__ a,
          _Float16* __restrict__ WhT, float* __restrict__ f1,
          float* __restrict__ f2)
{
  __shared__ _Float16 wlds[NHID * FIN];          // Wt[o][f], 64 KB
  const int gh = blockIdx.x;

  const float* Wg = W + (size_t)gh * FIN * NHID; // [512][64] row-major
  for (int i = threadIdx.x; i < FIN * NHID; i += blockDim.x) {
    int f = i >> 6, o = i & 63;
    wlds[o * FIN + f] = (_Float16)Wg[i];
  }
  __syncthreads();

  const int lane = threadIdx.x & 31;
  const int wave = threadIdx.x >> 5;
  const int hi   = lane >> 4;
  const int c16  = lane & 15;

  float a1c[4], a2c[4];
  #pragma unroll
  for (int t = 0; t < 4; ++t) {
    a1c[t] = a[gh * 2 * NHID + t * 16 + c16];
    a2c[t] = a[gh * 2 * NHID + NHID + t * 16 + c16];
  }

  for (int i = 0; i < 3; ++i) {
    const int ntile = blockIdx.y * 24 + wave * 3 + i;   // 8 blocks.y * 24 = 192
    const int nb = ntile * 16;
    const float* xrow = x + (size_t)(nb + c16) * FIN;

    v8f z = {0.f,0.f,0.f,0.f,0.f,0.f,0.f,0.f};
    v8f acc[4] = {z, z, z, z};

    for (int kb = 0; kb < FIN; kb += 32) {
      v16h av;
      #pragma unroll
      for (int j = 0; j < 16; ++j) {
        int k = kb + 16 * hi + j;
        av[j] = (_Float16)xrow[k];
      }
      #pragma unroll
      for (int ot = 0; ot < 4; ++ot) {
        const v16h* bp =
            (const v16h*)(wlds + (size_t)(ot * 16 + c16) * FIN + kb + 16 * hi);
        acc[ot] = wmma_f16(av, *bp, acc[ot]);
      }
    }

    // f1/f2: per-lane partial over the lane's 4 columns, butterfly over 16 lanes
    float p1[8], p2[8];
    #pragma unroll
    for (int r = 0; r < 8; ++r) { p1[r] = 0.f; p2[r] = 0.f; }
    #pragma unroll
    for (int t = 0; t < 4; ++t)
      #pragma unroll
      for (int r = 0; r < 8; ++r) {
        p1[r] += acc[t][r] * a1c[t];
        p2[r] += acc[t][r] * a2c[t];
      }
    #pragma unroll
    for (int r = 0; r < 8; ++r) {
      #pragma unroll
      for (int m = 1; m < 16; m <<= 1) {
        p1[r] += __shfl_xor(p1[r], m, 16);
        p2[r] += __shfl_xor(p2[r], m, 16);
      }
      if (c16 == 0) {
        f1[gh * NTOT + nb + r + 8 * hi] = p1[r];
        f2[gh * NTOT + nb + r + 8 * hi] = p2[r];
      }
    }

    // Wh transposed f16: WhT[gh][o][n]
    #pragma unroll
    for (int t = 0; t < 4; ++t)
      #pragma unroll
      for (int r = 0; r < 8; ++r)
        WhT[((size_t)gh * NHID + t * 16 + c16) * NTOT + nb + r + 8 * hi] =
            (_Float16)acc[t][r];
  }
}

// ---------------------------------------------------------------------------
// Kernel 2: flash-style masked attention + h = att@Wh + elu.
// One wave per (g,h, 32-row tile): two 16-row A tiles share every B load,
// halving L2 traffic for WhT. Streams m in 32-column steps.
// ---------------------------------------------------------------------------
__global__ void __launch_bounds__(32)
attn_kernel(const int* __restrict__ adj, const _Float16* __restrict__ WhT,
            const float* __restrict__ f1, const float* __restrict__ f2,
            float* __restrict__ hbuf)
{
  __shared__ float f2l[NTOT];                      // 12 KB
  const int gh = blockIdx.x;                       // 0..11
  const int g  = gh / NHEADS, h = gh % NHEADS;
  const int nb = blockIdx.y * 32;                  // 96 row-pair tiles
  const int lane = threadIdx.x;
  const int hi = lane >> 4, c16 = lane & 15;

  for (int i = lane; i < NTOT; i += 32) f2l[i] = f2[gh * NTOT + i];
  __syncthreads();

  const float f1n0 = f1[gh * NTOT + nb + c16];
  const float f1n1 = f1[gh * NTOT + nb + 16 + c16];
  const int* adjrow0 =
      adj + (size_t)g * NTOT * NTOT + (size_t)(nb + c16) * NTOT;
  const int* adjrow1 = adjrow0 + (size_t)16 * NTOT;
  const _Float16* whbase = WhT + (size_t)gh * NHID * NTOT;

  v8f z = {0.f,0.f,0.f,0.f,0.f,0.f,0.f,0.f};
  v8f acc0[4] = {z, z, z, z};
  v8f acc1[4] = {z, z, z, z};
  float lsum0 = 0.f, lsum1 = 0.f;

  for (int mb = 0; mb < NTOT; mb += 32) {
    v16h a0, a1;
    #pragma unroll
    for (int j = 0; j < 16; ++j) {
      int k = mb + 16 * hi + j;
      float f2v = f2l[k];
      int ad0 = adjrow0[k];
      int ad1 = adjrow1[k];
      float e0 = f1n0 + f2v;  e0 = e0 > 0.f ? e0 : ALPHAC * e0;
      float e1 = f1n1 + f2v;  e1 = e1 > 0.f ? e1 : ALPHAC * e1;
      float p0 = (ad0 > 0) ? __expf(e0) : 0.f;    // masked -> exp = 0
      float p1 = (ad1 > 0) ? __expf(e1) : 0.f;
      lsum0 += p0;  lsum1 += p1;
      a0[j] = (_Float16)p0;
      a1[j] = (_Float16)p1;
    }
    #pragma unroll
    for (int ot = 0; ot < 4; ++ot) {
      const v16h* bp = (const v16h*)(whbase +
          (size_t)(ot * 16 + c16) * NTOT + mb + 16 * hi);
      v16h bv = *bp;                               // shared by both row tiles
      acc0[ot] = wmma_f16(a0, bv, acc0[ot]);
      acc1[ot] = wmma_f16(a1, bv, acc1[ot]);
    }
  }

  // per-row softmax denominator: lanes l and l+16 hold row (l&15)'s halves
  float rs0 = lsum0 + __shfl_xor(lsum0, 16, 32);
  float rs1 = lsum1 + __shfl_xor(lsum1, 16, 32);

  #pragma unroll
  for (int r = 0; r < 8; ++r) {
    float rsr0 = __shfl(rs0, r + 8 * hi, 32);
    float rsr1 = __shfl(rs1, r + 8 * hi, 32);
    int n0 = nb + r + 8 * hi;
    int n1 = n0 + 16;
    #pragma unroll
    for (int ot = 0; ot < 4; ++ot) {
      float h0 = acc0[ot][r] / rsr0;
      h0 = h0 > 0.f ? h0 : (__expf(h0) - 1.f);     // elu
      hbuf[((size_t)g * NTOT + n0) * (NHEADS * NHID) + h * NHID + ot * 16 + c16]
          = h0;
      float h1 = acc1[ot][r] / rsr1;
      h1 = h1 > 0.f ? h1 : (__expf(h1) - 1.f);
      hbuf[((size_t)g * NTOT + n1) * (NHEADS * NHID) + h * NHID + ot * 16 + c16]
          = h1;
    }
  }
}

// ---------------------------------------------------------------------------
// Kernel 3: xi = elu(h@W_int^T + b_int); out = xi_flat@W_fus^T + b_fus;
// log_softmax. One wave per 16-row tile; WMMA for both small GEMMs.
// ---------------------------------------------------------------------------
__global__ void __launch_bounds__(32)
head_kernel(const float* __restrict__ hbuf, const float* __restrict__ W_int,
            const float* __restrict__ b_int, const float* __restrict__ W_fus,
            const float* __restrict__ b_fus, float* __restrict__ out)
{
  __shared__ float xi_lds[16 * 64];                 // 16 rows x 48 (padded) f32
  const int nb = blockIdx.x * 16;
  const int lane = threadIdx.x, hi = lane >> 4, c16 = lane & 15;
  v8f z = {0.f,0.f,0.f,0.f,0.f,0.f,0.f,0.f};

  for (int g = 0; g < NGRAPH; ++g) {
    v8f c = z;
    const float* hrow =
        hbuf + ((size_t)g * NTOT + nb + c16) * (NHEADS * NHID);
    for (int kb = 0; kb < NHEADS * NHID; kb += 32) {
      v16h av, bv;
      #pragma unroll
      for (int j = 0; j < 16; ++j) {
        int ka = kb + 16 * hi + j;
        av[j] = (_Float16)hrow[ka];
        int kbv = kb + 16 * hi + j;
        bv[j] = (_Float16)W_int[c16 * (NHEADS * NHID) + kbv];
      }
      c = wmma_f16(av, bv, c);
    }
    float bi = b_int[c16];
    #pragma unroll
    for (int r = 0; r < 8; ++r) {
      float v = c[r] + bi;
      v = v > 0.f ? v : (__expf(v) - 1.f);          // elu
      xi_lds[(r + 8 * hi) * 64 + g * 16 + c16] = v;
    }
  }
  __syncthreads();

  // fusion GEMM: K = 48 padded to 64
  v8f o = z;
  for (int kb = 0; kb < 64; kb += 32) {
    v16h av, bv;
    #pragma unroll
    for (int j = 0; j < 16; ++j) {
      int ka = kb + 16 * hi + j;
      av[j] = (ka < 48) ? (_Float16)xi_lds[c16 * 64 + ka] : (_Float16)0.0f;
      int kbv = kb + 16 * hi + j;
      bv[j] = (kbv < 48) ? (_Float16)W_fus[c16 * 48 + kbv] : (_Float16)0.0f;
    }
    o = wmma_f16(av, bv, o);
  }

  const float bf = b_fus[c16];
  #pragma unroll
  for (int r = 0; r < 8; ++r) {
    float t = o[r] + bf;
    float mx = t;
    #pragma unroll
    for (int m = 1; m < 16; m <<= 1) mx = fmaxf(mx, __shfl_xor(mx, m, 16));
    float sm = __expf(t - mx);
    #pragma unroll
    for (int m = 1; m < 16; m <<= 1) sm += __shfl_xor(sm, m, 16);
    out[(size_t)(nb + r + 8 * hi) * NCLASS + c16] = t - mx - __logf(sm);
  }
}

// ---------------------------------------------------------------------------
// Kernel 4: l1 loss = mean(|W_fus|) over 16*48 = 768 elements
// ---------------------------------------------------------------------------
__global__ void __launch_bounds__(32)
l1_kernel(const float* __restrict__ W_fus, float* __restrict__ out)
{
  float s = 0.f;
  for (int i = threadIdx.x; i < NCLASS * NCLASS * NGRAPH; i += 32)
    s += fabsf(W_fus[i]);
  #pragma unroll
  for (int m = 1; m < 32; m <<= 1) s += __shfl_xor(s, m, 32);
  if (threadIdx.x == 0)
    out[(size_t)NTOT * NCLASS] = s / (float)(NCLASS * NCLASS * NGRAPH);
}

// ---------------------------------------------------------------------------
extern "C" void kernel_launch(void* const* d_in, const int* in_sizes, int n_in,
                              void* d_out, int out_size, void* d_ws,
                              size_t ws_size, hipStream_t stream)
{
  (void)in_sizes; (void)n_in; (void)out_size; (void)ws_size;
  const float* x     = (const float*)d_in[0];
  const int*   adj   = (const int*)  d_in[1];
  const float* W     = (const float*)d_in[2];
  const float* a     = (const float*)d_in[3];
  const float* W_int = (const float*)d_in[4];
  const float* b_int = (const float*)d_in[5];
  const float* W_fus = (const float*)d_in[6];
  const float* b_fus = (const float*)d_in[7];
  float* out = (float*)d_out;

  char* ws = (char*)d_ws;
  const size_t whT_bytes = (size_t)GH * NHID * NTOT * sizeof(_Float16); // 4.5 MB
  const size_t f_bytes   = (size_t)GH * NTOT * sizeof(float);           // 144 KB
  _Float16* WhT = (_Float16*)ws;
  float* f1   = (float*)(ws + whT_bytes);
  float* f2   = (float*)(ws + whT_bytes + f_bytes);
  float* hbuf = (float*)(ws + whT_bytes + 2 * f_bytes);                 // 9 MB

  wh_kernel  <<<dim3(GH, 8),        256, 0, stream>>>(x, W, a, WhT, f1, f2);
  attn_kernel<<<dim3(GH, NTOT/32),  32,  0, stream>>>(adj, WhT, f1, f2, hbuf);
  head_kernel<<<NTOT/16,            32,  0, stream>>>(hbuf, W_int, b_int,
                                                      W_fus, b_fus, out);
  l1_kernel  <<<1,                  32,  0, stream>>>(W_fus, out);
}
